// ManyClassDecoder_78675210928790
// MI455X (gfx1250) — matrix operand order repaired and checked by reading, not compiled, not run on hardware
//
#include <hip/hip_runtime.h>

// ---------------------------------------------------------------------------
// Types for CDNA5 WMMA (wave32): D(f32 16x16) = A(bf16 16x32) x B(bf16 32x16) + C
// ---------------------------------------------------------------------------
typedef __attribute__((ext_vector_type(16))) __bf16 bf16x16;
typedef __attribute__((ext_vector_type(8)))  __bf16 bf16x8;
typedef __attribute__((ext_vector_type(8)))  float  floatx8;

static __device__ inline bf16x16 cat16(bf16x8 lo, bf16x8 hi) {
  return __builtin_shufflevector(lo, hi, 0,1,2,3,4,5,6,7,8,9,10,11,12,13,14,15);
}

// A fragment (16x32 bf16, M x K). Lane L holds row (L&15).
// kb = (L<16)?0:8.  v16 elems 0..7 = K[kb..kb+7], elems 8..15 = K[kb+16..kb+23].
static __device__ inline bf16x16 load_a_frag(const __bf16* row_k0, int kb) {
  const bf16x8 lo = *(const bf16x8*)(row_k0 + kb);
  const bf16x8 hi = *(const bf16x8*)(row_k0 + kb + 16);
  return cat16(lo, hi);
}

// B fragment (32x16 bf16, K x N). Lane L holds column (L&15).
// ks = (L<16)?0:16.  v16 elems 0..7 = K[ks/2..ks/2+7], elems 8..15 = K[ks/2+16..ks/2+23].
// Source layout: per-column contiguous K (i.e. row-major [n][k]).
static __device__ inline bf16x16 load_b_frag(const __bf16* col_k0, int ks) {
  const bf16x8 lo = *(const bf16x8*)(col_k0 + ks / 2);
  const bf16x8 hi = *(const bf16x8*)(col_k0 + ks / 2 + 16);
  return cat16(lo, hi);
}

#define WMMA_BF16(a, b, c) \
  __builtin_amdgcn_wmma_f32_16x16x32_bf16(false, (a), false, (b), (short)0, (c), false, false)

// ---------------------------------------------------------------------------
// Kernel 0: transpose + convert W[512][384] (f32) -> WT[384][512] (bf16)
// so that WMMA B-fragments are contiguous 16B loads.
// ---------------------------------------------------------------------------
__global__ __launch_bounds__(256) void wconv_kernel(const float* __restrict__ W,
                                                    __bf16* __restrict__ WT) {
  int i = blockIdx.x * 256 + threadIdx.x;
  if (i < 512 * 384) {
    int col = i >> 9;        // 0..383
    int k   = i & 511;       // 0..511
    WT[i] = (__bf16)W[(size_t)k * 384 + col];
  }
}

// ---------------------------------------------------------------------------
// Kernel 1: projection  OUT[b][h][row][64] (bf16) = (E[b][row][:] @ W + bias) * outscale
// grid: (rows/16, B), block: 256 (8 waves). 8 waves cover 24 column tiles.
// ---------------------------------------------------------------------------
__global__ __launch_bounds__(256) void proj_wmma_kernel(
    const float* __restrict__ E, int rows,
    const __bf16* __restrict__ WT,     // [384][512] bf16
    const float* __restrict__ bias,    // [384]
    float outscale,
    __bf16* __restrict__ OUT)          // ((b*6+h)*rows + r)*64 + d
{
  __shared__ __bf16 et[16 * 512];      // 16 KB: one 16-row tile of E, bf16
  const int b = blockIdx.y;
  const int rowbase = blockIdx.x << 4;
  const int tid = threadIdx.x;

  const float* Eb = E + ((size_t)b * rows + rowbase) * 512;   // tile is contiguous
  for (int i = tid; i < 16 * 512; i += 256) et[i] = (__bf16)Eb[i];
  __syncthreads();

  const int lane  = tid & 31, wave = tid >> 5;
  const int lo    = lane & 15;
  const int kb    = (lane < 16) ? 0 : 8;
  const int ks    = (lane < 16) ? 0 : 16;
  const int hioff = (lane >= 16) ? 8 : 0;
  const __bf16* arow = &et[lo * 512];

  for (int ct = wave; ct < 24; ct += 8) {          // uniform per wave (3 tiles each)
    const int col = ct * 16 + lo;
    floatx8 c;
    const float bv = bias[col];                    // bias depends on col only
#pragma unroll
    for (int i = 0; i < 8; ++i) c[i] = bv;

    const __bf16* wcol = WT + (size_t)col * 512;
#pragma unroll 2
    for (int kk = 0; kk < 16; ++kk) {              // K = 512 = 16 x 32
      bf16x16 a  = load_a_frag(arow + kk * 32, kb);
      bf16x16 bm = load_b_frag(wcol + kk * 32, ks);
      c = WMMA_BF16(a, bm, c);
    }

    const int h = col >> 6, d = col & 63;
    __bf16* op = OUT + (((size_t)b * 6 + h) * rows + rowbase + hioff) * 64 + d;
#pragma unroll
    for (int i = 0; i < 8; ++i)                    // vgpr i -> row (i + hioff)
      op[(size_t)i * 64] = (__bf16)(c[i] * outscale);
  }
}

// ---------------------------------------------------------------------------
// Kernel 2: attention + one-hot scatter + head mean + log
// grid: (M/16, B), block: 256 (8 waves). Per head: 2-pass online softmax over
// N=2048 (128 N-tiles, 2 WMMAs each), scatter exp/sum into LDS class buckets
// (stride 520 to stagger banks). Buckets accumulate over 6 heads -> /6 = mean.
// ---------------------------------------------------------------------------
__global__ __launch_bounds__(256) void attn_scatter_kernel(
    const __bf16* __restrict__ Qbf,    // [b][h][1024][64] (pre-scaled by 1/8)
    const __bf16* __restrict__ Kbf,    // [b][h][2048][64]
    const int*    __restrict__ targets,// [b][2048]
    float*        __restrict__ out)    // [1024][8][512]
{
  __shared__ float pm[16][128];        // per-(wave,lane) partial max
  __shared__ float ps[16][128];        // per-(wave,lane) partial sum
  __shared__ float mx[16];
  __shared__ float isum[16];
  __shared__ float buckets[16 * 520];  // 16 rows x 512 classes, padded stride

  const int b     = blockIdx.y;
  const int mbase = blockIdx.x << 4;
  const int tid   = threadIdx.x;
  const int lane  = tid & 31, wave = tid >> 5;
  const int lo    = lane & 15;
  const int hioff = (lane >= 16) ? 8 : 0;
  const int kb    = (lane < 16) ? 0 : 8;
  const int ks    = (lane < 16) ? 0 : 16;

  for (int i = tid; i < 16 * 520; i += 256) buckets[i] = 0.f;
  __syncthreads();

  for (int h = 0; h < 6; ++h) {
    const __bf16* Qb = Qbf + (((size_t)b * 6 + h) * 1024 + mbase) * 64;
    const __bf16* Kb = Kbf + (((size_t)b * 6 + h) * 2048) * 64;

    // A fragments live in registers for the whole head (K=64 -> two frags)
    const __bf16* qrow = Qb + lo * 64;
    const bf16x16 a0 = load_a_frag(qrow,      kb);
    const bf16x16 a1 = load_a_frag(qrow + 32, kb);

    // ---- pass A: online per-lane max/sum over this lane's (row, col) stream
    float ml[8], sl[8];
#pragma unroll
    for (int i = 0; i < 8; ++i) { ml[i] = -3.0e38f; sl[i] = 0.f; }

    for (int nt = wave; nt < 128; nt += 8) {       // uniform per wave
      const __bf16* krow = Kb + (size_t)(nt * 16 + lo) * 64;
      bf16x16 b0 = load_b_frag(krow,      ks);
      bf16x16 b1 = load_b_frag(krow + 32, ks);
      floatx8 c;
#pragma unroll
      for (int i = 0; i < 8; ++i) c[i] = 0.f;
      c = WMMA_BF16(a0, b0, c);
      c = WMMA_BF16(a1, b1, c);
#pragma unroll
      for (int i = 0; i < 8; ++i) {
        float x  = c[i];                           // scale already folded into Q
        float nm = fmaxf(ml[i], x);
        sl[i] = sl[i] * __expf(ml[i] - nm) + __expf(x - nm);
        ml[i] = nm;
      }
    }
    const int slot = wave * 16 + lo;
#pragma unroll
    for (int i = 0; i < 8; ++i) { pm[i + hioff][slot] = ml[i]; ps[i + hioff][slot] = sl[i]; }
    __syncthreads();

    if (tid < 16) {                                // merge 128 partials per row
      float M = -3.0e38f, S = 0.f;
      for (int j = 0; j < 128; ++j) {
        float m2 = pm[tid][j], s2 = ps[tid][j];
        float nm = fmaxf(M, m2);
        S = S * __expf(M - nm) + s2 * __expf(m2 - nm);
        M = nm;
      }
      mx[tid]   = M;
      isum[tid] = 1.0f / S;
    }
    __syncthreads();

    float mxr[8], isr[8];
#pragma unroll
    for (int i = 0; i < 8; ++i) { mxr[i] = mx[i + hioff]; isr[i] = isum[i + hioff]; }

    // ---- pass B: recompute scores, scatter attention into class buckets
    for (int nt = wave; nt < 128; nt += 8) {
      const __bf16* krow = Kb + (size_t)(nt * 16 + lo) * 64;
      bf16x16 b0 = load_b_frag(krow,      ks);
      bf16x16 b1 = load_b_frag(krow + 32, ks);
      floatx8 c;
#pragma unroll
      for (int i = 0; i < 8; ++i) c[i] = 0.f;
      c = WMMA_BF16(a0, b0, c);
      c = WMMA_BF16(a1, b1, c);
      const int cls = targets[b * 2048 + nt * 16 + lo];   // class of this column
#pragma unroll
      for (int i = 0; i < 8; ++i) {
        float a = __expf(c[i] - mxr[i]) * isr[i];
        atomicAdd(&buckets[(i + hioff) * 520 + cls], a);  // ds_add_f32
      }
    }
    __syncthreads();
  }

  // ---- head mean, clip, log, write [M, B, C]
  const float inv6 = 1.0f / 6.0f;
  for (int i = tid; i < 16 * 512; i += 256) {
    int r = i >> 9, c = i & 511;
    float v = fmaxf(buckets[r * 520 + c] * inv6, 1e-5f);
    out[(((size_t)(mbase + r)) * 8 + b) * 512 + c] = __logf(v + 3e-5f);
  }
}

// ---------------------------------------------------------------------------
// Launch. Workspace layout (bytes):
//   [0,          393216)   WqT bf16 [384][512]
//   [393216,     786432)   WkT bf16 [384][512]
//   [786432,    7077888)   Qbf bf16 [8][6][1024][64]   (pre-scaled by 1/8)
//   [7077888,  19660800)   Kbf bf16 [8][6][2048][64]
// Requires ws_size >= 19660800 (~18.75 MB).
// ---------------------------------------------------------------------------
extern "C" void kernel_launch(void* const* d_in, const int* in_sizes, int n_in,
                              void* d_out, int out_size, void* d_ws, size_t ws_size,
                              hipStream_t stream) {
  const float* train   = (const float*)d_in[0];   // [8][2048][512]
  const float* test    = (const float*)d_in[1];   // [8][1024][512]
  const int*   targets = (const int*)  d_in[2];   // [8][2048]
  const float* Wq      = (const float*)d_in[3];   // [512][384]
  const float* bq      = (const float*)d_in[4];   // [384]
  const float* Wk      = (const float*)d_in[5];   // [512][384]
  const float* bk      = (const float*)d_in[6];   // [384]
  float* out = (float*)d_out;                     // [1024][8][512]

  char* ws = (char*)d_ws;
  __bf16* WqT = (__bf16*)(ws);
  __bf16* WkT = (__bf16*)(ws + 393216);
  __bf16* Qbf = (__bf16*)(ws + 786432);
  __bf16* Kbf = (__bf16*)(ws + 7077888);

  // weight convert/transpose: 196608 elems each
  wconv_kernel<<<768, 256, 0, stream>>>(Wq, WqT);
  wconv_kernel<<<768, 256, 0, stream>>>(Wk, WkT);

  // projections (softmax scale 1/sqrt(64)=0.125 folded into Q)
  proj_wmma_kernel<<<dim3(64, 8),  256, 0, stream>>>(test,  1024, WqT, bq, 0.125f, Qbf);
  proj_wmma_kernel<<<dim3(128, 8), 256, 0, stream>>>(train, 2048, WkT, bk, 1.0f,   Kbf);

  // attention + class scatter + mean + log
  attn_scatter_kernel<<<dim3(64, 8), 256, 0, stream>>>(Qbf, Kbf, targets, out);
}
